// UNetRecurrentIncr_17575006175566
// MI455X (gfx1250) — matrix operand order repaired and checked by reading, not compiled, not run on hardware
//
#include <hip/hip_runtime.h>
#include <math.h>

// ---------------------------------------------------------------------------
// CDNA5 (gfx1250) recurrent U-Net forward.
// Conv layers = implicit GEMM on v_wmma_f32_16x16x32_bf16 (wave32).
// Block tile: M=MT (32/64 Cout) x N=128 pixels x K=64 per LDS round.
// Weights pre-converted to bf16; A tile DMA'd to LDS by the Tensor Data Mover.
// ---------------------------------------------------------------------------

typedef __attribute__((ext_vector_type(16))) __bf16 v16bf;
typedef __attribute__((ext_vector_type(8)))  __bf16 v8bf;
typedef __attribute__((ext_vector_type(8)))  float  v8f;
typedef __attribute__((ext_vector_type(4)))  unsigned int v4u;
typedef __attribute__((ext_vector_type(8)))  int v8i;
typedef __attribute__((ext_vector_type(4)))  int v4i;

#define BATCH 4
#define TILE_PIX 128   // pixels per block (8 waves x 16)
#define ASTRIDE 64     // sA row stride (bf16)
#define BSTRIDE 72     // sB row stride (bf16): 64 k + 8 pad, 144B rows -> 16B aligned

#ifndef USE_TDM
#if defined(__has_builtin)
#if __has_builtin(__builtin_amdgcn_tensor_load_to_lds)
#define USE_TDM 1
#endif
#endif
#endif
#ifndef USE_TDM
#define USE_TDM 0
#endif

struct ConvDesc {
    const float*  x0;   // first input  [B, c0, Hin, Win]
    const float*  x1;   // second input [B, Cin-c0, Hin, Win] (channel concat), may be null
    const __bf16* wb;   // [Cout, Cin*K*K] pre-converted bf16 weights
    const float*  bias; // [Cout]
    const float*  skip; // optional pre-activation add, layout of y, may be null
    float* y;           // [B, Cout, Hout, Wout]
    int c0, Cin, Cout, Hin, Win, Hout, Wout, K, stride, pad, act; // act: 0 none, 1 relu
};

template <int MT>
__launch_bounds__(256)
__global__ void conv_wmma_kernel(ConvDesc d) {
    const int tid  = threadIdx.x;
    const int lane = tid & 31;
    const int wave = tid >> 5;          // 0..7
    const int hi   = lane >> 4;         // 0 = lanes 0-15, 1 = lanes 16-31
    const int lm   = lane & 15;

    const int mBase   = blockIdx.y * MT;        // Cout tile
    const int pixBase = blockIdx.x * TILE_PIX;  // flat (b, oy, ox) pixel tile

    const int HWout  = d.Hout * d.Wout;
    const int Ntot   = BATCH * HWout;
    const int K      = d.K;
    const int KK     = K * K;
    const int CinKK  = d.Cin * KK;
    const int ksteps = (CinKK + 63) >> 6;
    const int HWin   = d.Hin * d.Win;
    const int c1n    = d.Cin - d.c0;
    const int c0HW   = d.c0 * HWin;

    __shared__ __align__(16) __bf16 sA[MT * ASTRIDE];       // [m][k]
    __shared__ __align__(16) __bf16 sB[TILE_PIX * BSTRIDE]; // [n][k]

    // one-time zero init of sA (TDM overwrites each round; keeps reads defined)
    #pragma unroll
    for (int i = tid; i < MT * ASTRIDE; i += 256) sA[i] = (__bf16)0.f;

    // ---- B-fill assignment: this thread owns pixel nFill, k in [kOff, kOff+32) ----
    const int nFill = tid & 127;
    const int kOff  = (tid >> 7) << 5;  // 0 or 32
    const int pFill = pixBase + nFill;
    int bimg = 0, oy = 0, ox = 0;
    const bool pValid = pFill < Ntot;
    if (pValid) {
        bimg = pFill / HWout;
        int r = pFill - bimg * HWout;
        oy = r / d.Wout;
        ox = r - oy * d.Wout;
    }
    const int iy0 = oy * d.stride - d.pad;
    const int ix0 = ox * d.stride - d.pad;
    const float* x0b = d.x0 + (long)bimg * c0HW;                  // [c0, Hin, Win]
    const float* x1b = d.x1 ? d.x1 + (long)bimg * c1n * HWin : d.x1;

#if USE_TDM
    const unsigned ldsA = (unsigned)(size_t)&sA[0];
    const unsigned long long wTileBase =
        (unsigned long long)(size_t)(d.wb + (long)mBase * CinKK);
#endif

    v8f acc[MT / 16];
    #pragma unroll
    for (int s = 0; s < MT / 16; ++s) acc[s] = (v8f){};

    for (int ks = 0; ks < ksteps; ++ks) {
        const int kBase = ks << 6;
        __syncthreads();   // previous round's fragment reads complete

#if USE_TDM
        // ---- stage A via Tensor Data Mover: MT x 64 bf16 tile, row stride CinKK ----
        if (wave == 0) {
            const unsigned long long ga = wTileBase + (unsigned long long)kBase * 2ull;
            const unsigned remK = (unsigned)(CinKK - kBase);   // remaining extent -> OOB zero-fill
            const unsigned remM = (unsigned)(d.Cout - mBase);
            v4u g0;
            g0.x = 1u;                                          // count=1, user D#
            g0.y = ldsA;                                        // LDS dest (bytes)
            g0.z = (unsigned)(ga & 0xFFFFFFFFu);                // global_addr[31:0]
            g0.w = (unsigned)((ga >> 32) & 0x1FFFFFFu) | (2u << 30); // addr[56:32] | type=2
            v8i g1;
            g1[0] = (int)(1u << 16);                            // data_size=1 -> 2 bytes
            g1[1] = (int)((remK & 0xFFFFu) << 16);              // tensor_dim0[15:0]
            g1[2] = (int)((remK >> 16) & 0xFFFFu) |             // tensor_dim0[31:16]
                    (int)((remM & 0xFFFFu) << 16);              // tensor_dim1[15:0]
            g1[3] = (int)((remM >> 16) & 0xFFFFu) |             // tensor_dim1[31:16]
                    (int)(64u << 16);                           // tile_dim0 = 64
            g1[4] = MT;                                         // tile_dim1 = MT, tile_dim2 = 0
            g1[5] = CinKK;                                      // tensor_dim0_stride[31:0]
            g1[6] = 0;
            g1[7] = 0;
            const v4i z4 = {0, 0, 0, 0};
            const v8i z8 = {0, 0, 0, 0, 0, 0, 0, 0};
            __builtin_amdgcn_tensor_load_to_lds(g0, g1, z4, z4, z8, 0);
        }
#else
        // ---- fallback: manual bf16 weight staging ----
        for (int i = tid; i < MT * 64; i += 256) {
            const int m = i >> 6, k = i & 63;
            const int kidx = kBase + k;
            __bf16 v = (__bf16)0.f;
            if (kidx < CinKK) v = d.wb[(long)(mBase + m) * CinKK + kidx];
            sA[m * ASTRIDE + k] = v;
        }
#endif

        // ---- stage B (im2col): pixel nFill, 32 consecutive k; one div per round ----
        {
            const int kidx0 = kBase + kOff;
            int cin = kidx0 / KK;
            int rem = kidx0 - cin * KK;
            int kh  = rem / K;
            int kw  = rem - kh * K;
            int iy  = iy0 + kh;
            int ix  = ix0 + kw;
            int off = cin * HWin + iy * d.Win + ix;
            #pragma unroll 4
            for (int j = 0; j < 32; ++j) {
                float v = 0.f;
                if (pValid && cin < d.Cin &&
                    iy >= 0 && iy < d.Hin && ix >= 0 && ix < d.Win) {
                    v = (cin < d.c0) ? x0b[off] : x1b[off - c0HW];
                }
                sB[nFill * BSTRIDE + kOff + j] = (__bf16)v;
                // mixed-radix advance of (cin, kh, kw) with incremental address
                ++kw; ++ix; ++off;
                if (kw == K) {
                    kw = 0; ++kh; ix -= K; ++iy; off += d.Win - K;
                    if (kh == K) {
                        kh = 0; ++cin; iy = iy0; ix = ix0;
                        off = cin * HWin + iy0 * d.Win + ix0;
                    }
                }
            }
        }

#if USE_TDM
        if (wave == 0) __builtin_amdgcn_s_wait_tensorcnt(0);
#endif
        __syncthreads();

        // ---- 2 K-chunks x (MT/16) M-subtiles of WMMA ----
        const int nRow = (wave * 16 + lm) * BSTRIDE;
        #pragma unroll
        for (int c = 0; c < 2; ++c) {
            const int kc = c << 5;
            // B fragment: col = wave*16+lm; lanes 0-15 k=kc+0..7,16..23, lanes 16-31 k=kc+8..15,24..31
            const int boff = nRow + kc + (hi ? 8 : 0);
            const v8bf blo = *(const v8bf*)&sB[boff];
            const v8bf bhi = *(const v8bf*)&sB[boff + 16];
            const v16bf bfrag = __builtin_shufflevector(blo, bhi,
                0, 1, 2, 3, 4, 5, 6, 7, 8, 9, 10, 11, 12, 13, 14, 15);
            #pragma unroll
            for (int s = 0; s < MT / 16; ++s) {
                const int aoff = (s * 16 + lm) * ASTRIDE + kc + (hi ? 8 : 0);
                const v8bf alo = *(const v8bf*)&sA[aoff];
                const v8bf ahi = *(const v8bf*)&sA[aoff + 16];
                const v16bf afrag = __builtin_shufflevector(alo, ahi,
                    0, 1, 2, 3, 4, 5, 6, 7, 8, 9, 10, 11, 12, 13, 14, 15);
                acc[s] = __builtin_amdgcn_wmma_f32_16x16x32_bf16(
                    false, afrag, false, bfrag, (short)0, acc[s], false, false);
            }
        }
    }

    // ---- store D (f32 16x16 layout): VGPR r -> M = r + 8*hi, N = lm ----
    const int pOut = pixBase + wave * 16 + lm;
    if (pOut < Ntot) {
        const int bo = pOut / HWout;
        const int r  = pOut - bo * HWout;
        const long outBase = (long)bo * d.Cout * HWout + r;
        #pragma unroll
        for (int s = 0; s < MT / 16; ++s) {
            #pragma unroll
            for (int rr = 0; rr < 8; ++rr) {
                const int cout = mBase + s * 16 + rr + (hi ? 8 : 0);
                if (cout < d.Cout) {
                    float v = acc[s][rr] + d.bias[cout];
                    const long oidx = outBase + (long)cout * HWout;
                    if (d.skip) v += d.skip[oidx];
                    if (d.act == 1) v = fmaxf(v, 0.f);
                    d.y[oidx] = v;
                }
            }
        }
    }
}

// ---------------------------------------------------------------------------
// Elementwise kernels
// ---------------------------------------------------------------------------

__global__ void f32_to_bf16_kernel(const float* __restrict__ s, __bf16* __restrict__ t, int n) {
    const int i = blockIdx.x * 256 + threadIdx.x;
    if (i < n) t[i] = (__bf16)s[i];
}

__device__ __forceinline__ float sigmoidf_(float x) { return 1.f / (1.f + __expf(-x)); }

__global__ void lstm_pw_kernel(const float* __restrict__ gates,
                               const float* __restrict__ cprev,
                               float* __restrict__ hout, float* __restrict__ cout_,
                               int C, int HW, int total) {
    const int idx = blockIdx.x * 256 + threadIdx.x;
    if (idx >= total) return;
    const int chw = C * HW;
    const int b  = idx / chw;
    const int r  = idx - b * chw;
    const int ch = r / HW;
    const int px = r - ch * HW;
    const long g0 = (long)b * 4 * chw + (long)ch * HW + px;
    const float ig = sigmoidf_(gates[g0]);
    const float fg = sigmoidf_(gates[g0 + (long)C * HW]);
    const float og = sigmoidf_(gates[g0 + (long)2 * C * HW]);
    const float gg = tanhf(gates[g0 + (long)3 * C * HW]);
    const float cn = fg * cprev[idx] + ig * gg;
    cout_[idx] = cn;
    hout[idx]  = og * tanhf(cn);
}

// bilinear 2x upsample (half-pixel centers) of (a + b)
__global__ void up2x_add_kernel(const float* __restrict__ a, const float* __restrict__ b,
                                float* __restrict__ y, int Hin, int Win, int total) {
    const int idx = blockIdx.x * 256 + threadIdx.x;
    if (idx >= total) return;
    const int Hout = Hin * 2, Wout = Win * 2;
    const int hwo = Hout * Wout;
    const int pc = idx / hwo;          // fused (batch*channel) plane
    const int r  = idx - pc * hwo;
    const int oy = r / Wout;
    const int ox = r - oy * Wout;
    const float fy = (oy + 0.5f) * 0.5f - 0.5f;
    const float fx = (ox + 0.5f) * 0.5f - 0.5f;
    int y0 = (int)floorf(fy), x0 = (int)floorf(fx);
    const float wy = fy - (float)y0, wx = fx - (float)x0;
    int y1 = y0 + 1, x1 = x0 + 1;
    y0 = min(max(y0, 0), Hin - 1); y1 = min(max(y1, 0), Hin - 1);
    x0 = min(max(x0, 0), Win - 1); x1 = min(max(x1, 0), Win - 1);
    const float* ap = a + (long)pc * Hin * Win;
    const float* bp = b + (long)pc * Hin * Win;
    const int o00 = y0 * Win + x0, o01 = y0 * Win + x1;
    const int o10 = y1 * Win + x0, o11 = y1 * Win + x1;
    const float s00 = ap[o00] + bp[o00], s01 = ap[o01] + bp[o01];
    const float s10 = ap[o10] + bp[o10], s11 = ap[o11] + bp[o11];
    y[idx] = (1.f - wy) * ((1.f - wx) * s00 + wx * s01) +
             wy         * ((1.f - wx) * s10 + wx * s11);
}

// 1x1 conv (32 -> 1) over (cur + head), sigmoid
__global__ void pred_kernel(const float* __restrict__ cur, const float* __restrict__ head,
                            const float* __restrict__ w, const float* __restrict__ bias,
                            float* __restrict__ img, int HW, int total) {
    const int idx = blockIdx.x * 256 + threadIdx.x;
    if (idx >= total) return;
    const int b  = idx / HW;
    const int px = idx - b * HW;
    float acc = bias[0];
    #pragma unroll
    for (int ch = 0; ch < 32; ++ch) {
        const long o = (long)(b * 32 + ch) * HW + px;
        acc += (cur[o] + head[o]) * w[ch];
    }
    img[idx] = sigmoidf_(acc);
}

// ---------------------------------------------------------------------------
// Host-side sequencing
// ---------------------------------------------------------------------------

static void launch_conv(hipStream_t s,
                        const float* x0, const float* x1, int c0,
                        const __bf16* wb, const float* bias, const float* skip, float* y,
                        int Cin, int Cout, int Hin, int Win, int Hout, int Wout,
                        int K, int stride, int pad, int act) {
    ConvDesc d;
    d.x0 = x0; d.x1 = x1; d.wb = wb; d.bias = bias; d.skip = skip; d.y = y;
    d.c0 = c0; d.Cin = Cin; d.Cout = Cout;
    d.Hin = Hin; d.Win = Win; d.Hout = Hout; d.Wout = Wout;
    d.K = K; d.stride = stride; d.pad = pad; d.act = act;
    const int Ntot = BATCH * Hout * Wout;
    const int gx = (Ntot + TILE_PIX - 1) / TILE_PIX;
    if ((Cout & 63) == 0) {
        hipLaunchKernelGGL((conv_wmma_kernel<64>), dim3(gx, Cout / 64), dim3(256), 0, s, d);
    } else {
        hipLaunchKernelGGL((conv_wmma_kernel<32>), dim3(gx, (Cout + 31) / 32), dim3(256), 0, s, d);
    }
}

extern "C" void kernel_launch(void* const* d_in, const int* in_sizes, int n_in,
                              void* d_out, int out_size, void* d_ws, size_t ws_size,
                              hipStream_t stream) {
    (void)in_sizes; (void)n_in; (void)out_size; (void)ws_size;

    const float* x   = (const float*)d_in[0];
    const float* h0i = (const float*)d_in[1];
    const float* c0i = (const float*)d_in[2];
    const float* h1i = (const float*)d_in[3];
    const float* c1i = (const float*)d_in[4];
    const float* h2i = (const float*)d_in[5];
    const float* c2i = (const float*)d_in[6];
    const float* head_b = (const float*)d_in[8];
    const float* pred_w = (const float*)d_in[9];
    const float* pred_b = (const float*)d_in[10];
    const float* enc0_b = (const float*)d_in[12];
    const float* lstm0_b = (const float*)d_in[14];
    const float* enc1_b = (const float*)d_in[16];
    const float* lstm1_b = (const float*)d_in[18];
    const float* enc2_b = (const float*)d_in[20];
    const float* lstm2_b = (const float*)d_in[22];
    const float* res0_b1 = (const float*)d_in[24];
    const float* res0_b2 = (const float*)d_in[26];
    const float* res1_b1 = (const float*)d_in[28];
    const float* res1_b2 = (const float*)d_in[30];
    const float* dec0_b = (const float*)d_in[32];
    const float* dec1_b = (const float*)d_in[34];
    const float* dec2_b = (const float*)d_in[36];

    float* out = (float*)d_out;
    float* img = out;                       // 4*1*256*256   = 262144
    float* h0o = out + 262144;              // 4*64*128*128  = 4194304
    float* c0o = h0o + 4194304;
    float* h1o = c0o + 4194304;             // 4*128*64*64   = 2097152
    float* c1o = h1o + 2097152;
    float* h2o = c1o + 2097152;             // 4*256*32*32   = 1048576
    float* c2o = h2o + 1048576;

    // workspace arena (floats), with buffer reuse across the timeline
    float* ws   = (float*)d_ws;
    float* head = ws;                       // 4*32*256*256 = 8388608 (persistent)
    float* bufA = head + 8388608;           // 16777216 : gates0, up0, up2
    float* bufB = bufA + 16777216;          //  8388608 : gates1, up1, dec2
    float* bufC = bufB + 8388608;           //  4194304 : enc0, dec0, dec1
    float* bufD = bufC + 4194304;           //  2097152 : enc1, res1 out
    float* bufE = bufD + 2097152;           //  1048576 : enc2, res mid
    float* bufF = bufE + 1048576;           //  1048576 : res0 out
    __bf16* wbf = (__bf16*)(bufF + 1048576);// bf16 weight arena (~10.7M elems)

    // ---- pre-convert all conv weights to bf16 ----
    long wo = 0;
    auto cvt_w = [&](int srcIdx, int n) -> const __bf16* {
        __bf16* dst = wbf + wo;
        hipLaunchKernelGGL(f32_to_bf16_kernel, dim3((n + 255) / 256), dim3(256), 0, stream,
                           (const float*)d_in[srcIdx], dst, n);
        wo += n;
        return dst;
    };
    const __bf16* head_wb = cvt_w(7, 32 * 5 * 25);
    const __bf16* enc0_wb = cvt_w(11, 64 * 32 * 25);
    const __bf16* lstm0_wb = cvt_w(13, 256 * 128 * 9);
    const __bf16* enc1_wb = cvt_w(15, 128 * 64 * 25);
    const __bf16* lstm1_wb = cvt_w(17, 512 * 256 * 9);
    const __bf16* enc2_wb = cvt_w(19, 256 * 128 * 25);
    const __bf16* lstm2_wb = cvt_w(21, 1024 * 512 * 9);
    const __bf16* res0_w1b = cvt_w(23, 256 * 256 * 9);
    const __bf16* res0_w2b = cvt_w(25, 256 * 256 * 9);
    const __bf16* res1_w1b = cvt_w(27, 256 * 256 * 9);
    const __bf16* res1_w2b = cvt_w(29, 256 * 256 * 9);
    const __bf16* dec0_wb = cvt_w(31, 128 * 256 * 25);
    const __bf16* dec1_wb = cvt_w(33, 64 * 128 * 25);
    const __bf16* dec2_wb = cvt_w(35, 32 * 64 * 25);

    // 1) head: 5 -> 32, k5 s1 p2, relu, 256x256
    launch_conv(stream, x, nullptr, 5, head_wb, head_b, nullptr, head,
                5, 32, 256, 256, 256, 256, 5, 1, 2, 1);
    // 2) enc0: 32 -> 64, k5 s2 p2, relu, -> 128x128
    launch_conv(stream, head, nullptr, 32, enc0_wb, enc0_b, nullptr, bufC,
                32, 64, 256, 256, 128, 128, 5, 2, 2, 1);
    // 3) lstm0 gates: concat(enc0, h0) 128 -> 256, k3 p1
    launch_conv(stream, bufC, h0i, 64, lstm0_wb, lstm0_b, nullptr, bufA,
                128, 256, 128, 128, 128, 128, 3, 1, 1, 0);
    // 4) lstm0 pointwise
    {
        const int total = 4 * 64 * 128 * 128;
        hipLaunchKernelGGL(lstm_pw_kernel, dim3((total + 255) / 256), dim3(256), 0, stream,
                           bufA, c0i, h0o, c0o, 64, 128 * 128, total);
    }
    // 5) enc1: 64 -> 128, k5 s2 p2, relu, -> 64x64
    launch_conv(stream, h0o, nullptr, 64, enc1_wb, enc1_b, nullptr, bufD,
                64, 128, 128, 128, 64, 64, 5, 2, 2, 1);
    // 6) lstm1 gates: concat(enc1, h1) 256 -> 512, k3 p1
    launch_conv(stream, bufD, h1i, 128, lstm1_wb, lstm1_b, nullptr, bufB,
                256, 512, 64, 64, 64, 64, 3, 1, 1, 0);
    // 7) lstm1 pointwise
    {
        const int total = 4 * 128 * 64 * 64;
        hipLaunchKernelGGL(lstm_pw_kernel, dim3((total + 255) / 256), dim3(256), 0, stream,
                           bufB, c1i, h1o, c1o, 128, 64 * 64, total);
    }
    // 8) enc2: 128 -> 256, k5 s2 p2, relu, -> 32x32
    launch_conv(stream, h1o, nullptr, 128, enc2_wb, enc2_b, nullptr, bufE,
                128, 256, 64, 64, 32, 32, 5, 2, 2, 1);
    // 9) lstm2 gates: concat(enc2, h2) 512 -> 1024, k3 p1
    launch_conv(stream, bufE, h2i, 256, lstm2_wb, lstm2_b, nullptr, bufA,
                512, 1024, 32, 32, 32, 32, 3, 1, 1, 0);
    // 10) lstm2 pointwise
    {
        const int total = 4 * 256 * 32 * 32;
        hipLaunchKernelGGL(lstm_pw_kernel, dim3((total + 255) / 256), dim3(256), 0, stream,
                           bufA, c2i, h2o, c2o, 256, 32 * 32, total);
    }
    // 11) res0: relu(conv1); relu(conv2 + skip)
    launch_conv(stream, h2o, nullptr, 256, res0_w1b, res0_b1, nullptr, bufE,
                256, 256, 32, 32, 32, 32, 3, 1, 1, 1);
    launch_conv(stream, bufE, nullptr, 256, res0_w2b, res0_b2, h2o, bufF,
                256, 256, 32, 32, 32, 32, 3, 1, 1, 1);
    // 12) res1
    launch_conv(stream, bufF, nullptr, 256, res1_w1b, res1_b1, nullptr, bufE,
                256, 256, 32, 32, 32, 32, 3, 1, 1, 1);
    launch_conv(stream, bufE, nullptr, 256, res1_w2b, res1_b2, bufF, bufD,
                256, 256, 32, 32, 32, 32, 3, 1, 1, 1);
    // 13) up0 = up2x(res_out + h2), 256ch 32->64
    {
        const int total = 4 * 256 * 64 * 64;
        hipLaunchKernelGGL(up2x_add_kernel, dim3((total + 255) / 256), dim3(256), 0, stream,
                           bufD, h2o, bufA, 32, 32, total);
    }
    // 14) dec0: 256 -> 128, k5 p2, relu, 64x64
    launch_conv(stream, bufA, nullptr, 256, dec0_wb, dec0_b, nullptr, bufC,
                256, 128, 64, 64, 64, 64, 5, 1, 2, 1);
    // 15) up1 = up2x(dec0 + h1), 128ch 64->128
    {
        const int total = 4 * 128 * 128 * 128;
        hipLaunchKernelGGL(up2x_add_kernel, dim3((total + 255) / 256), dim3(256), 0, stream,
                           bufC, h1o, bufB, 64, 64, total);
    }
    // 16) dec1: 128 -> 64, k5 p2, relu, 128x128
    launch_conv(stream, bufB, nullptr, 128, dec1_wb, dec1_b, nullptr, bufC,
                128, 64, 128, 128, 128, 128, 5, 1, 2, 1);
    // 17) up2 = up2x(dec1 + h0), 64ch 128->256
    {
        const int total = 4 * 64 * 256 * 256;
        hipLaunchKernelGGL(up2x_add_kernel, dim3((total + 255) / 256), dim3(256), 0, stream,
                           bufC, h0o, bufA, 128, 128, total);
    }
    // 18) dec2: 64 -> 32, k5 p2, relu, 256x256
    launch_conv(stream, bufA, nullptr, 64, dec2_wb, dec2_b, nullptr, bufB,
                64, 32, 256, 256, 256, 256, 5, 1, 2, 1);
    // 19) pred: sigmoid(1x1 conv over (dec2 + head))
    {
        const int total = 4 * 256 * 256;
        hipLaunchKernelGGL(pred_kernel, dim3((total + 255) / 256), dim3(256), 0, stream,
                           bufB, head, pred_w, pred_b, img, 256 * 256, total);
    }
}
